// blockSA_69724499083450
// MI455X (gfx1250) — matrix operand order repaired and checked by reading, not compiled, not run on hardware
//
#include <hip/hip_runtime.h>

// ---------------------------------------------------------------------------
// blockSA on gfx1250 (MI455X).
//
// Mathematical simplification of the reference:
//   _t2tSA is an exact identity: softmax over j sums to 1 and multiplies a
//   term independent of j.  Hence h = xb, o = v, and G*o+(1-G)*v = v.
//   The whole module reduces to a blockwise (r=3) s2tSA:
//     Y1 = relu(x @ W1^T + b1)          (3072 x 256)
//     Z  = Y1 @ W^T + b                 (3072 x 256)
//     out[g,h] = sum_k softmax_k(Z[3g+k,h]) * x[3g+k,h],  g in [0,1024)
//
// v2: operands are pre-converted f32->f16 ONCE (cvt kernel), so the GEMM
// inner loop is pure global_load_b128 + v_wmma_f32_16x16x32_f16 with no
// per-iteration v_cvt work.  Each wave computes a 32x64 strip (2 m-tiles x
// 4 n-tiles) to amortize fragment loads across 8 WMMAs per k-step.
// ---------------------------------------------------------------------------

typedef _Float16 v16h __attribute__((ext_vector_type(16)));
typedef _Float16 v8h  __attribute__((ext_vector_type(8)));
typedef _Float16 v4h  __attribute__((ext_vector_type(4)));
typedef float    v8f  __attribute__((ext_vector_type(8)));
typedef float    v4f  __attribute__((ext_vector_type(4)));

#define HDIM            256      // hidden size H
#define M_ROWS          3072     // B*S = 2*1536
#define WAVES_PER_BLOCK 8
#define MT_PER_WAVE     2        // 2 m-tiles of 16 -> 32 rows per wave
#define NT_PER_WAVE     4        // 4 n-tiles of 16 -> 64 cols per wave

// ---------------------------------------------------------------------------
// f32 -> f16 elementwise conversion (n must be a multiple of 4)
// ---------------------------------------------------------------------------
__global__ __launch_bounds__(256) void cvt_f32_f16(const float* __restrict__ src,
                                                   _Float16* __restrict__ dst,
                                                   int n) {
  const int i = (blockIdx.x * blockDim.x + threadIdx.x) * 4;
  if (i >= n) return;
  const v4f v = *reinterpret_cast<const v4f*>(src + i);
  v4h h;
#pragma unroll
  for (int e = 0; e < 4; ++e) h[e] = (_Float16)v[e];
  *reinterpret_cast<v4h*>(dst + i) = h;
}

// ---------------------------------------------------------------------------
// C[m,n] = act( sum_k A[m,k] * W[n,k] + bias[n] )
// A:(M,H) f16 row-major, W:(H,H) f16 row-major, bias f32.
// OUT_F16: store C as f16 (feeds next GEMM); else f32.
// ---------------------------------------------------------------------------
template <bool RELU, bool OUT_F16>
__global__ __launch_bounds__(256) void gemm_xWT(const _Float16* __restrict__ A,
                                                const _Float16* __restrict__ W,
                                                const float* __restrict__ bias,
                                                void* __restrict__ Cout) {
  const int lane = threadIdx.x & 31;
  const int wave = threadIdx.x >> 5;
  const int flat = blockIdx.x * WAVES_PER_BLOCK + wave;

  const int n_groups = HDIM / (16 * NT_PER_WAVE);   // 4
  const int m_super  = flat / n_groups;             // 0..95
  const int n_base   = (flat % n_groups) * (16 * NT_PER_WAVE);
  const int m_base   = m_super * (16 * MT_PER_WAVE);

  const int rlo = lane & 15;    // row/col within tile
  const int hi  = lane >> 4;    // half-wave select (0/1)

  v8f acc[MT_PER_WAVE][NT_PER_WAVE] = {};   // f32 accumulators

  for (int k0 = 0; k0 < HDIM; k0 += 32) {
    // ---- A fragments (16x32 f16), contiguous k = 16*hi + e (same order as B) ----
    // lane<16 : elems 0..15 = K k0..k0+15
    // lane>=16: elems 0..15 = K k0+16..k0+31
    v16h afrag[MT_PER_WAVE];
#pragma unroll
    for (int s = 0; s < MT_PER_WAVE; ++s) {
      const _Float16* pa = A + (m_base + s * 16 + rlo) * HDIM + k0 + hi * 16;
      const v8h alo = *reinterpret_cast<const v8h*>(pa);
      const v8h ahi = *reinterpret_cast<const v8h*>(pa + 8);
#pragma unroll
      for (int e = 0; e < 8; ++e) {
        afrag[s][e]     = alo[e];
        afrag[s][8 + e] = ahi[e];
      }
    }

#pragma unroll
    for (int t = 0; t < NT_PER_WAVE; ++t) {
      // ---- B fragment (32x16 f16): B[k][n] = W[n][k] ----
      // lane holds col n = n_tile + (lane&15); 16 consecutive K from k0+hi*16
      const int n = n_base + t * 16 + rlo;
      const v16h bfrag =
          *reinterpret_cast<const v16h*>(W + n * HDIM + k0 + hi * 16);
#pragma unroll
      for (int s = 0; s < MT_PER_WAVE; ++s) {
        acc[s][t] = __builtin_amdgcn_wmma_f32_16x16x32_f16(
            /*neg_a=*/false, afrag[s], /*neg_b=*/false, bfrag,
            /*c_mod=*/(short)0, acc[s][t], /*reuse_a=*/false, /*reuse_b=*/false);
      }
    }
  }

  // ---- epilogue: C/D layout: elem v, lane -> (M = v + 8*hi, N = lane&15) ----
#pragma unroll
  for (int s = 0; s < MT_PER_WAVE; ++s) {
#pragma unroll
    for (int t = 0; t < NT_PER_WAVE; ++t) {
      const int n  = n_base + t * 16 + rlo;
      const float bn = bias[n];
#pragma unroll
      for (int v = 0; v < 8; ++v) {
        float val = acc[s][t][v] + bn;
        if (RELU) val = fmaxf(val, 0.0f);
        const int m = m_base + s * 16 + v + 8 * hi;
        if (OUT_F16)
          ((_Float16*)Cout)[m * HDIM + n] = (_Float16)val;
        else
          ((float*)Cout)[m * HDIM + n] = val;
      }
    }
  }
}

// ---------------------------------------------------------------------------
// out[g,h] = sum_k softmax_k(Z[3g+k,h]) * X[3g+k,h]
// ---------------------------------------------------------------------------
__global__ __launch_bounds__(256) void pool3(const float* __restrict__ Z,
                                             const float* __restrict__ X,
                                             float* __restrict__ out) {
  const int idx = blockIdx.x * blockDim.x + threadIdx.x;  // g*H + h
  const int g = idx >> 8;        // H == 256
  const int h = idx & (HDIM - 1);
  if (g >= M_ROWS / 3) return;

  const float* z = Z + (3 * g) * HDIM + h;
  const float* x = X + (3 * g) * HDIM + h;
  const float z0 = z[0], z1 = z[HDIM], z2 = z[2 * HDIM];
  const float m  = fmaxf(z0, fmaxf(z1, z2));
  const float e0 = __expf(z0 - m);
  const float e1 = __expf(z1 - m);
  const float e2 = __expf(z2 - m);
  const float inv = 1.0f / (e0 + e1 + e2);
  out[idx] = (e0 * x[0] + e1 * x[HDIM] + e2 * x[2 * HDIM]) * inv;
}

// ---------------------------------------------------------------------------
extern "C" void kernel_launch(void* const* d_in, const int* in_sizes, int n_in,
                              void* d_out, int out_size, void* d_ws,
                              size_t ws_size, hipStream_t stream) {
  (void)in_sizes; (void)n_in; (void)out_size; (void)ws_size;
  const float* x  = (const float*)d_in[0];   // (2,1536,256)
  const float* W1 = (const float*)d_in[1];   // s2t_W1_w (H,H)
  const float* b1 = (const float*)d_in[2];   // s2t_W1_b (H)
  const float* Wc = (const float*)d_in[3];   // s2t_W_w  (H,H)
  const float* bc = (const float*)d_in[4];   // s2t_W_b  (H)
  // remaining inputs (m_*, g_*) are mathematically inert — see header comment.

  const size_t NX = (size_t)M_ROWS * HDIM;   // 786432
  const size_t NW = (size_t)HDIM * HDIM;     // 65536

  // workspace layout (f16 region first, then f32 Z)
  _Float16* xh  = (_Float16*)d_ws;           // x   in f16
  _Float16* W1h = xh + NX;                   // W1  in f16
  _Float16* Wch = W1h + NW;                  // W   in f16
  _Float16* Y1h = Wch + NW;                  // relu(x@W1^T+b1) in f16
  float*    Zs  = (float*)(Y1h + NX);        // logits in f32 (4B aligned)

  dim3 blk(256);
  cvt_f32_f16<<<(int)(NX / 4 / 256), blk, 0, stream>>>(x,  xh,  (int)NX);
  cvt_f32_f16<<<(int)(NW / 4 / 256), blk, 0, stream>>>(W1, W1h, (int)NW);
  cvt_f32_f16<<<(int)(NW / 4 / 256), blk, 0, stream>>>(Wc, Wch, (int)NW);

  const int gemm_blocks = (M_ROWS / (16 * MT_PER_WAVE)) *
                          (HDIM / (16 * NT_PER_WAVE)) / WAVES_PER_BLOCK;  // 48

  gemm_xWT<true,  true ><<<gemm_blocks, blk, 0, stream>>>(xh,  W1h, b1, Y1h);
  gemm_xWT<false, false><<<gemm_blocks, blk, 0, stream>>>(Y1h, Wch, bc, Zs);

  const int pool_elems  = (M_ROWS / 3) * HDIM;            // 1024*256
  const int pool_blocks = (pool_elems + 255) / 256;       // 1024
  pool3<<<pool_blocks, blk, 0, stream>>>(Zs, x, (float*)d_out);
}
